// SegmentalLM_6691559047534
// MI455X (gfx1250) — matrix operand order repaired and checked by reading, not compiled, not run on hardware
//
#include <hip/hip_runtime.h>
#include <stdint.h>

// ---------------- problem constants ----------------
#define V_    4000
#define VP    4096          // vocab padded for 128-wide WMMA tiling
#define H_    256
#define M_    128
#define B_    32
#define L_    4
#define S_    126           // M-2 segments
#define G4H   1024          // 4*H
#define KCAT  512           // H (x) + H (h)
#define ROWS_DEC 4032       // S_*B_
#define NROWS (5*ROWS_DEC)  // 20160 decoder-output rows
#define LOGINF 1000000.0f

// ---------------- WMMA plumbing (gfx1250, wave32) ----------------
typedef __bf16 bf16x16 __attribute__((ext_vector_type(16)));
typedef float  f32x8   __attribute__((ext_vector_type(8)));
typedef unsigned int uintx4 __attribute__((ext_vector_type(4)));
typedef unsigned short u16;
typedef __attribute__((address_space(1))) const uintx4 guintx4;  // force global_load_b128

union Frag16 { bf16x16 v; uintx4 q[2]; };

__device__ __forceinline__ u16 f2bf(float f) {
  unsigned u = __float_as_uint(f);
  unsigned r = u + 0x7FFFu + ((u >> 16) & 1u);   // round-to-nearest-even
  return (u16)(r >> 16);
}
__device__ __forceinline__ float bf2f(u16 h) { return __uint_as_float(((unsigned)h) << 16); }

#define LOG2E 1.4426950408889634f
__device__ __forceinline__ float fast_exp(float x) { return __builtin_amdgcn_exp2f(x * LOG2E); }
__device__ __forceinline__ float fast_rcp(float x) { return __builtin_amdgcn_rcpf(x); }
// branch-free, saturating at +-1 / [0,1]; v_exp + v_rcp + fma, no exec-mask divergence
__device__ __forceinline__ float sigf(float x)   { return fast_rcp(1.f + fast_exp(-x)); }
__device__ __forceinline__ float tanh_f(float x) { return 1.f - 2.f * fast_rcp(1.f + fast_exp(2.f * x)); }

// Launder a pointer so the compiler cannot hoist/CSE loads through it across
// loop iterations (prevents B-fragment LICM -> 1000+ VGPR live set -> scratch spills).
__device__ __forceinline__ const u16* launder(const u16* p) {
  asm volatile("" : "+v"(p));
  return p;
}

__device__ __forceinline__ f32x8 wmma_bf16(bf16x16 a, bf16x16 b, f32x8 c) {
  // D = A(16x32 bf16) * B(32x16 bf16) + C(16x16 f32)
  return __builtin_amdgcn_wmma_f32_16x16x32_bf16(false, a, false, b, (short)0, c, false, false);
}

// A fragment: 16x32 tile at (m0,k0) of row-major [rows x lda] bf16 matrix.
// Per ISA 7.12.2: lane m=lane&15; lanes<16 hold k {0..7,16..23}, lanes>=16 {8..15,24..31}.
// Source may be LDS or global -> keep generic addressing.
__device__ __forceinline__ bf16x16 load_a(const u16* base, int lda, int m0, int k0, int lane) {
  int m = m0 + (lane & 15);
  int khb = (lane >> 4) << 3;                    // 0 or 8
  const u16* p = base + m * lda + k0 + khb;
  Frag16 f;
  f.q[0] = *(const uintx4*)(p);                  // k = k0+khb+0..7
  f.q[1] = *(const uintx4*)(p + 16);             // k = k0+16+khb+0..7
  return f.v;
}
// B fragment: 32x16 tile at (k0,n0); source stored row-major as W[n][k] ([N x ldk]).
// lane column n=n0+(lane&15); lanes<16 hold k {0..7,16..23}, lanes>=16 {8..15,24..31}.
// Source is always global memory -> address_space(1) => global_load_b128 (LOADcnt only).
__device__ __forceinline__ bf16x16 load_b(const u16* base, int ldk, int n0, int k0, int lane) {
  int n = n0 + (lane & 15);
  int k = k0 + ((lane >> 4) << 3);
  const u16* p = base + n * ldk + k;             // two 16-byte chunks
  Frag16 f;
  f.q[0] = *(guintx4*)(unsigned long long)(p);
  f.q[1] = *(guintx4*)(unsigned long long)(p + 16);
  return f.v;
}

// ---------------- prep kernels ----------------
__global__ void prep_emb_kernel(const float* __restrict__ emb, const float* __restrict__ e2vb_in,
                                u16* __restrict__ emb_bf, float* __restrict__ e2vb) {
  int idx = blockIdx.x * blockDim.x + threadIdx.x;
  if (idx < VP * H_) {
    int v = idx >> 8, c = idx & 255;
    emb_bf[idx] = (v < V_) ? f2bf(emb[v * H_ + c]) : (u16)0;
  }
  if (idx < VP) e2vb[idx] = (idx < V_) ? e2vb_in[idx] : -LOGINF;
}

__global__ void prep_wcat_kernel(const float* __restrict__ Wih, const float* __restrict__ Whh,
                                 const float* __restrict__ bih, const float* __restrict__ bhh,
                                 u16* __restrict__ wcat, float* __restrict__ bias) {
  int idx = blockIdx.x * blockDim.x + threadIdx.x;
  if (idx < G4H * KCAT) {
    int n = idx >> 9, k = idx & 511;
    float v = (k < H_) ? Wih[n * H_ + k] : Whh[n * H_ + (k - H_)];
    wcat[idx] = f2bf(v);
  }
  if (idx < G4H) bias[idx] = bih[idx] + bhh[idx];
}

__global__ void prep_mat_kernel(const float* __restrict__ W, u16* __restrict__ out, int n) {
  int idx = blockIdx.x * blockDim.x + threadIdx.x;
  if (idx < n) out[idx] = f2bf(W[idx]);
}

// inputs_bf: (M_+L_, B_, H_) bf16, rows >= M_ zero-padded
__global__ void embed_kernel(const int* __restrict__ x, const float* __restrict__ emb,
                             u16* __restrict__ inputs_bf) {
  int idx = blockIdx.x * blockDim.x + threadIdx.x;
  if (idx >= (M_ + L_) * B_ * H_) return;
  int c = idx & 255;
  int r = idx >> 8;
  int b = r & 31;
  int m = r >> 5;
  if (m < M_) {
    int tok = x[b * M_ + m];
    inputs_bf[idx] = f2bf(emb[tok * H_ + c]);
  } else {
    inputs_bf[idx] = 0;
  }
}

// ---------------- encoder LSTM: 1 block, 32 waves, 128 sequential steps ----------------
// wave w: mt = w>>4 (batch 16-row half), jt = w&15 (16 hidden cols); owns 4 gate tiles + c-frag.
__global__ __launch_bounds__(1024) void encoder_kernel(
    const u16* __restrict__ inputs_bf, const u16* __restrict__ wcat,
    const float* __restrict__ bias, const float* __restrict__ h0,
    const float* __restrict__ c0, u16* __restrict__ enc_out_bf) {
  extern __shared__ char smem[];
  u16* acat = (u16*)smem;                        // 32 x 512 bf16: [x | h]
  int tid = threadIdx.x;
  int w = tid >> 5, lane = tid & 31;
  int mt = w >> 4, jt = w & 15;
  int jc = jt * 16 + (lane & 15);
  int mrow = mt * 16 + ((lane >> 4) << 3);

  for (int i = tid; i < B_ * H_; i += 1024) {    // h init (broadcast h0)
    int b = i >> 8, c = i & 255;
    acat[b * KCAT + H_ + c] = f2bf(h0[c]);
  }
  float bi = bias[jc], bff = bias[256 + jc], bg = bias[512 + jc], bo = bias[768 + jc];
  float c0v = c0[jc];
  float cfr[8];
  for (int r = 0; r < 8; ++r) cfr[r] = c0v;

#pragma unroll 1
  for (int t = 0; t < M_; ++t) {
    {                                            // stage x_t into LDS (16B per thread)
      int e0 = tid * 8, b = e0 >> 8, c = e0 & 255;
      *(uintx4*)(&acat[b * KCAT + c]) = *(const uintx4*)(inputs_bf + t * (B_ * H_) + e0);
    }
    __syncthreads();
    const u16* wc = launder(wcat);               // keep weight loads inside this step
    f32x8 acc[4];
    for (int g = 0; g < 4; ++g) for (int r = 0; r < 8; ++r) acc[g][r] = 0.f;
#pragma unroll 1
    for (int k0 = 0; k0 < KCAT; k0 += 32) {
      bf16x16 af = load_a(acat, KCAT, mt * 16, k0, lane);
      for (int g = 0; g < 4; ++g)
        acc[g] = wmma_bf16(af, load_b(wc, KCAT, g * H_ + jt * 16, k0, lane), acc[g]);
    }
    __syncthreads();                             // all reads of acat done before h overwrite
    for (int r = 0; r < 8; ++r) {
      float iv = sigf(acc[0][r] + bi);
      float fv = sigf(acc[1][r] + bff);
      float gv = tanh_f(acc[2][r] + bg);
      float ov = sigf(acc[3][r] + bo);
      float cn = fv * cfr[r] + iv * gv;
      cfr[r] = cn;
      float h = ov * tanh_f(cn);
      int m = mrow + r;
      acat[m * KCAT + H_ + jc] = f2bf(h);        // recurrent h
      enc_out_bf[(t * B_ + m) * H_ + jc] = f2bf(0.5f * h);  // enc_out = 0.5*h
    }
  }
}

// ---------------- sos / dec_h0 projections ----------------
__global__ __launch_bounds__(256) void proj_kernel(
    const u16* __restrict__ enc_out_bf, const u16* __restrict__ wbf_sos,
    const u16* __restrict__ wbf_dht, const float* __restrict__ sos_b,
    const float* __restrict__ dht_b, u16* __restrict__ sos_bf, u16* __restrict__ dh0_bf) {
  int s = blockIdx.x;
  int kind = blockIdx.y;                         // 0 = sos, 1 = dht(tanh)
  const u16* A = enc_out_bf + s * (B_ * H_);     // enc_prev = enc_out[s]
  const u16* W = kind ? wbf_dht : wbf_sos;
  const float* bias = kind ? dht_b : sos_b;
  u16* dst = kind ? dh0_bf : sos_bf;
  int tid = threadIdx.x, w = tid >> 5, lane = tid & 31;
  for (int i = 0; i < 4; ++i) {
    int tile = w * 4 + i;
    int mt = tile >> 4, nt = tile & 15;
    f32x8 acc; for (int r = 0; r < 8; ++r) acc[r] = 0.f;
    for (int k0 = 0; k0 < H_; k0 += 32)
      acc = wmma_bf16(load_a(A, H_, mt * 16, k0, lane), load_b(W, H_, nt * 16, k0, lane), acc);
    int n = nt * 16 + (lane & 15);
    float bv = bias[n];
    int mbase = mt * 16 + ((lane >> 4) << 3);
    for (int r = 0; r < 8; ++r) {
      float v = acc[r] + bv;
      if (kind) v = tanh_f(v);
      dst[(s * B_ + mbase + r) * H_ + n] = f2bf(v);
    }
  }
}

// ---------------- decoder LSTM: 126 blocks x 32 waves, 5 sequential steps ----------------
__global__ __launch_bounds__(1024) void decoder_kernel(
    const u16* __restrict__ inputs_bf, const u16* __restrict__ sos_bf,
    const u16* __restrict__ dh0_bf, const u16* __restrict__ wcat,
    const float* __restrict__ bias, u16* __restrict__ dec_out_bf) {
  extern __shared__ char smem[];
  u16* acat = (u16*)smem;
  int s = blockIdx.x;
  int tid = threadIdx.x, w = tid >> 5, lane = tid & 31;
  int mt = w >> 4, jt = w & 15;
  int jc = jt * 16 + (lane & 15);
  int mrow = mt * 16 + ((lane >> 4) << 3);
  {                                              // h init = dec_h0[s]
    int e0 = tid * 8, b = e0 >> 8, c = e0 & 255;
    *(uintx4*)(&acat[b * KCAT + H_ + c]) = *(const uintx4*)(dh0_bf + s * (B_ * H_) + e0);
  }
  float bi = bias[jc], bff = bias[256 + jc], bg = bias[512 + jc], bo = bias[768 + jc];
  float cfr[8];
  for (int r = 0; r < 8; ++r) cfr[r] = 0.f;      // dec_c0 = 0

#pragma unroll 1
  for (int t = 0; t < L_ + 1; ++t) {
    const u16* xs = (t == 0) ? (sos_bf + s * (B_ * H_)) : (inputs_bf + (s + t) * (B_ * H_));
    {
      int e0 = tid * 8, b = e0 >> 8, c = e0 & 255;
      *(uintx4*)(&acat[b * KCAT + c]) = *(const uintx4*)(xs + e0);
    }
    __syncthreads();
    const u16* wc = launder(wcat);               // keep weight loads inside this step
    f32x8 acc[4];
    for (int g = 0; g < 4; ++g) for (int r = 0; r < 8; ++r) acc[g][r] = 0.f;
#pragma unroll 1
    for (int k0 = 0; k0 < KCAT; k0 += 32) {
      bf16x16 af = load_a(acat, KCAT, mt * 16, k0, lane);
      for (int g = 0; g < 4; ++g)
        acc[g] = wmma_bf16(af, load_b(wc, KCAT, g * H_ + jt * 16, k0, lane), acc[g]);
    }
    __syncthreads();
    for (int r = 0; r < 8; ++r) {
      float iv = sigf(acc[0][r] + bi);
      float fv = sigf(acc[1][r] + bff);
      float gv = tanh_f(acc[2][r] + bg);
      float ov = sigf(acc[3][r] + bo);
      float cn = fv * cfr[r] + iv * gv;
      cfr[r] = cn;
      float h = ov * tanh_f(cn);
      int m = mrow + r;
      acat[m * KCAT + H_ + jc] = f2bf(h);
      dec_out_bf[((t * S_ + s) * B_ + m) * H_ + jc] = f2bf(h);
    }
  }
}

// ---------------- fused logits GEMM + logsumexp (logits stay in LDS) ----------------
// 1260 blocks of 256 threads; each block: 16 rows x 4096 padded vocab in 256KB LDS.
__global__ __launch_bounds__(256) void lse_kernel(
    const u16* __restrict__ dec_out_bf, const u16* __restrict__ emb_bf,
    const float* __restrict__ e2vb, float* __restrict__ lse) {
  extern __shared__ char smem[];
  float* slog = (float*)smem;                    // 16 * 4096 f32
  __shared__ float red[16][17];
  int tid = threadIdx.x, w = tid >> 5, lane = tid & 31;
  int r0 = blockIdx.x * 16;
  const u16* A = dec_out_bf + r0 * H_;
  bf16x16 afr[8];
  for (int kk = 0; kk < 8; ++kk) afr[kk] = load_a(A, H_, 0, kk * 32, lane);
  int lcol = lane & 15;
  int mbase = (lane >> 4) << 3;
#pragma unroll 1
  for (int ch = 0; ch < 32; ++ch) {              // 8 waves x 16 cols = 128 vocab / chunk
    int n0 = ch * 128 + w * 16;
    const u16* eb = launder(emb_bf);
    f32x8 acc; for (int r = 0; r < 8; ++r) acc[r] = 0.f;
    for (int kk = 0; kk < 8; ++kk)
      acc = wmma_bf16(afr[kk], load_b(eb, H_, n0, kk * 32, lane), acc);
    float bb = e2vb[n0 + lcol];
    for (int r = 0; r < 8; ++r)
      slog[(mbase + r) * VP + n0 + lcol] = acc[r] + bb;
  }
  __syncthreads();
  int row = tid >> 4, seg = tid & 15;
  const float* p = slog + row * VP + seg * 256;
  float mx = -3.0e38f;
  for (int j = 0; j < 256; ++j) mx = fmaxf(mx, p[j]);
  red[row][seg] = mx;
  __syncthreads();
  if (seg == 0) {
    float m = red[row][0];
    for (int j = 1; j < 16; ++j) m = fmaxf(m, red[row][j]);
    red[row][16] = m;
  }
  __syncthreads();
  float rm = red[row][16];
  float sum = 0.f;
  for (int j = 0; j < 256; ++j) sum += __expf(p[j] - rm);
  red[row][seg] = sum;
  __syncthreads();
  if (seg == 0) {
    float tt = 0.f;
    for (int j = 0; j < 16; ++j) tt += red[row][j];
    lse[r0 + row] = rm + __logf(tt);
  }
}

// ---------------- per-segment log-prob rows (gather dots + cumsum + eos + mask) ----------------
__global__ void logpy_kernel(const int* __restrict__ x, const u16* __restrict__ dec_out_bf,
                             const u16* __restrict__ emb_bf, const float* __restrict__ e2vb,
                             const float* __restrict__ lse, float* __restrict__ logpy) {
  int s = blockIdx.x;                            // 0..125, start = s+1
  int b = threadIdx.x;                           // 0..31
  int tok1 = x[b * M_ + (s + 1)];
  float is_start = (tok1 <= 2) ? -LOGINF : 0.f;  // PUNC/ENG/NUM
  int jlen = (126 - s < 4) ? (126 - s) : 4;      // min(L, M-1-start)
  float cum = 0.f;
  float eosb = e2vb[3];
  const u16* ee = emb_bf + 3 * H_;
  for (int k = 0; k < 4; ++k) {
    int pos = s + 1 + k;
    int tok = (pos < M_) ? x[b * M_ + pos] : 0;
    float isv = (pos < M_) ? ((tok <= 2) ? -LOGINF : 0.f) : 0.f;
    int rowT = k * ROWS_DEC + s * B_ + b;
    const u16* dr = dec_out_bf + rowT * H_;
    const u16* er = emb_bf + tok * H_;
    float td = 0.f;
    for (int j = 0; j < H_; ++j) td += bf2f(dr[j]) * bf2f(er[j]);
    float tlp = td + e2vb[tok] - lse[rowT];
    int rowE = (k + 1) * ROWS_DEC + s * B_ + b;
    const u16* de = dec_out_bf + rowE * H_;
    float ed = 0.f;
    for (int j = 0; j < H_; ++j) ed += bf2f(de[j]) * bf2f(ee[j]);
    float eosv = ed + eosb - lse[rowE];
    cum += tlp + ((k >= 1) ? isv : 0.f);
    float lp = cum + ((k >= 1) ? is_start : 0.f) + eosv;
    if (k >= jlen) lp = -LOGINF;
    logpy[((s + 1) * 4 + k) * B_ + b] = lp;      // rows 1..126; row 0 never read
  }
}

// ---------------- DP over segmentations + final reduction ----------------
__global__ void dp_kernel(const float* __restrict__ logpy, const int* __restrict__ lengths,
                          float* __restrict__ out) {
  int b = threadIdx.x;                           // lane == batch
  float buf[4] = {0.f, -LOGINF, -LOGINF, -LOGINF};
  int te = lengths[b] - 2;
  float nll = 0.f;                               // alphas[0] = 0
  for (int e = 1; e <= 126; ++e) {
    float v[4];
    for (int k = 0; k < 4; ++k) {
      int idx = e - k;
      float sl = (idx >= 1) ? logpy[(idx * 4 + k) * B_ + b] : -LOGINF;
      v[k] = buf[k] + sl;
    }
    float m = fmaxf(fmaxf(v[0], v[1]), fmaxf(v[2], v[3]));
    float a = m + __logf(__expf(v[0] - m) + __expf(v[1] - m) + __expf(v[2] - m) + __expf(v[3] - m));
    buf[3] = buf[2]; buf[2] = buf[1]; buf[1] = buf[0]; buf[0] = a;
    if (e == te) nll = -a;
  }
  float tot = nll;
  float lensum = (float)lengths[b];
  for (int off = 16; off >= 1; off >>= 1) {
    tot += __shfl_down(tot, off, 32);
    lensum += __shfl_down(lensum, off, 32);
  }
  if (b == 0) out[0] = tot / (lensum - 2.f * (float)B_);
}

// ---------------- host orchestration ----------------
extern "C" void kernel_launch(void* const* d_in, const int* in_sizes, int n_in,
                              void* d_out, int out_size, void* d_ws, size_t ws_size,
                              hipStream_t stream) {
  (void)in_sizes; (void)n_in; (void)out_size; (void)ws_size;
  const int*   x       = (const int*)  d_in[0];
  const int*   lengths = (const int*)  d_in[1];
  const float* emb     = (const float*)d_in[2];
  const float* e2vb_in = (const float*)d_in[3];
  const float* encWih  = (const float*)d_in[4];
  const float* encWhh  = (const float*)d_in[5];
  const float* encbih  = (const float*)d_in[6];
  const float* encbhh  = (const float*)d_in[7];
  const float* enc_h0  = (const float*)d_in[8];
  const float* enc_c0  = (const float*)d_in[9];
  const float* decWih  = (const float*)d_in[10];
  const float* decWhh  = (const float*)d_in[11];
  const float* decbih  = (const float*)d_in[12];
  const float* decbhh  = (const float*)d_in[13];
  const float* dht_W   = (const float*)d_in[14];
  const float* dht_b   = (const float*)d_in[15];
  const float* sos_W   = (const float*)d_in[16];
  const float* sos_b   = (const float*)d_in[17];
  float* out = (float*)d_out;

  char* ws = (char*)d_ws;
  size_t off = 0;
  auto carve = [&](size_t bytes) -> char* {
    char* p = ws + off;
    off += (bytes + 255) & ~(size_t)255;
    return p;
  };
  u16*   emb_bf     = (u16*)  carve((size_t)VP * H_ * 2);
  float* e2vb       = (float*)carve((size_t)VP * 4);
  u16*   wcat_enc   = (u16*)  carve((size_t)G4H * KCAT * 2);
  u16*   wcat_dec   = (u16*)  carve((size_t)G4H * KCAT * 2);
  float* bias_enc   = (float*)carve((size_t)G4H * 4);
  float* bias_dec   = (float*)carve((size_t)G4H * 4);
  u16*   sosW_bf    = (u16*)  carve((size_t)H_ * H_ * 2);
  u16*   dhtW_bf    = (u16*)  carve((size_t)H_ * H_ * 2);
  u16*   inputs_bf  = (u16*)  carve((size_t)(M_ + L_) * B_ * H_ * 2);
  u16*   enc_out_bf = (u16*)  carve((size_t)M_ * B_ * H_ * 2);
  u16*   sos_bf     = (u16*)  carve((size_t)S_ * B_ * H_ * 2);
  u16*   dh0_bf     = (u16*)  carve((size_t)S_ * B_ * H_ * 2);
  u16*   dec_out_bf = (u16*)  carve((size_t)NROWS * H_ * 2);
  float* lse        = (float*)carve((size_t)NROWS * 4);
  float* logpy      = (float*)carve((size_t)(S_ + 1) * 4 * B_ * 4);

  // prep
  prep_emb_kernel<<<(VP * H_ + 255) / 256, 256, 0, stream>>>(emb, e2vb_in, emb_bf, e2vb);
  prep_wcat_kernel<<<(G4H * KCAT + 255) / 256, 256, 0, stream>>>(encWih, encWhh, encbih, encbhh, wcat_enc, bias_enc);
  prep_wcat_kernel<<<(G4H * KCAT + 255) / 256, 256, 0, stream>>>(decWih, decWhh, decbih, decbhh, wcat_dec, bias_dec);
  prep_mat_kernel<<<(H_ * H_ + 255) / 256, 256, 0, stream>>>(sos_W, sosW_bf, H_ * H_);
  prep_mat_kernel<<<(H_ * H_ + 255) / 256, 256, 0, stream>>>(dht_W, dhtW_bf, H_ * H_);
  embed_kernel<<<((M_ + L_) * B_ * H_ + 255) / 256, 256, 0, stream>>>(x, emb, inputs_bf);

  // encoder (sequential, single WGP) -> projections -> decoder
  encoder_kernel<<<1, 1024, B_ * KCAT * 2, stream>>>(inputs_bf, wcat_enc, bias_enc, enc_h0, enc_c0, enc_out_bf);
  proj_kernel<<<dim3(S_, 2), 256, 0, stream>>>(enc_out_bf, sosW_bf, dhtW_bf, sos_b, dht_b, sos_bf, dh0_bf);
  decoder_kernel<<<S_, 1024, B_ * KCAT * 2, stream>>>(inputs_bf, sos_bf, dh0_bf, wcat_dec, bias_dec, dec_out_bf);

  // fused logits + logsumexp (256KB LDS of logits per block, never hits HBM)
  lse_kernel<<<NROWS / 16, 256, 16 * VP * 4, stream>>>(dec_out_bf, emb_bf, e2vb, lse);

  // segment log-probs and DP
  logpy_kernel<<<S_, B_, 0, stream>>>(x, dec_out_bf, emb_bf, e2vb, lse, logpy);
  dp_kernel<<<1, B_, 0, stream>>>(logpy, lengths, out);
}
